// AttentionPairBias_7456063225923
// MI455X (gfx1250) — matrix operand order repaired and checked by reading, not compiled, not run on hardware
//
#include <hip/hip_runtime.h>
#include <hip/hip_bf16.h>

#define Bb 2
#define II 768
#define CA 768
#define CS 384
#define CZ 128
#define HH 16
#define DD 48
#define HCd 768
#define MROWS (Bb * II)   // 1536
#define EPSf 1e-5f

typedef unsigned short u16;
typedef __attribute__((ext_vector_type(16))) __bf16 v16bf;
typedef __attribute__((ext_vector_type(8)))  float  v8f;

// ---------- bf16 <-> f32 helpers ----------
__device__ __forceinline__ u16 f2bfu(float f) {
  union { float f; unsigned u; } x; x.f = f;
  unsigned r = (x.u + 0x7FFFu + ((x.u >> 16) & 1u)) >> 16;
  return (u16)r;
}
__device__ __forceinline__ float bfu2f(u16 s) {
  union { unsigned u; float f; } x; x.u = ((unsigned)s) << 16;
  return x.f;
}
__device__ __forceinline__ __bf16 u2b(u16 s) {
  union { u16 u; __bf16 b; } c; c.u = s; return c.b;
}
__device__ __forceinline__ float sigf(float x) { return 1.0f / (1.0f + __expf(-x)); }

// ---------- fragment-ready ("swizzled") tiled layouts ----------
// Both A (16Mx32K) and B (32Kx16N) tiles are stored as 512 u16:
//   tile_base + lane*16 + elem   -- lane's whole fragment is 32 contiguous bytes.
// A-swizzle: element i of lane L (m=L&15, half=L>>4):
//   i<8 : k%32 = half*8 + i ; i>=8 : k%32 = 16 + half*8 + (i-8)
// B-swizzle: element i of lane L (n=L&15, half=L>>4): same k order as A-swizzle
__device__ __forceinline__ size_t aswz_idx(int m, int k, int K) {
  int kc = k & 31;
  int half = (kc >> 3) & 1;
  int run  = kc >> 4;
  int idx  = kc & 7;
  int lane = (m & 15) + 16 * half;
  return ((size_t)(m >> 4) * (K >> 5) + (size_t)(k >> 5)) * 512 + lane * 16 + run * 8 + idx;
}
// load a full fragment (A or B) from a swizzled tile in GLOBAL memory: 2x b128
__device__ __forceinline__ v16bf load_frag_sw(const u16* tile_base) {
  const uint4* p = (const uint4*)(tile_base + (size_t)(threadIdx.x & 31) * 16);
  union { uint4 q[2]; v16bf b; } u;
  u.q[0] = p[0]; u.q[1] = p[1];
  return u.b;
}
// A fragment from row-major LDS [m][k], stride ld (pairs/runs contiguous -> ds_load_b128)
__device__ __forceinline__ v16bf load_frag_a_lds(const u16* a, int ld) {
  int lane = threadIdx.x & 31;
  int m = lane & 15, half = lane >> 4;
  v16bf f;
#pragma unroll
  for (int v = 0; v < 8; ++v) {
    int kb = (v < 4 ? 0 : 16) + half * 8 + (v & 3) * 2;
    f[2 * v]     = u2b(a[m * ld + kb]);
    f[2 * v + 1] = u2b(a[m * ld + kb + 1]);
  }
  return f;
}
__device__ __forceinline__ v8f wmma_bf(v16bf a, v16bf b, v8f c) {
  return __builtin_amdgcn_wmma_f32_16x16x32_bf16(false, a, false, b, (short)0, c, false, false);
}

// ================= weight f32 -> bf16 + B-swizzle =================
// one thread per (tile,lane): reads 16 strided f32, writes 32 contiguous bytes
__global__ void k_cvt_bswz(const float* __restrict__ w, u16* __restrict__ out,
                           int K, int N, int n_units) {
  int tid = blockIdx.x * blockDim.x + threadIdx.x;
  if (tid >= n_units) return;
  int tile = tid >> 5, lane = tid & 31;
  int kt = tile % (K >> 5), nt = tile / (K >> 5);
  int n  = nt * 16 + (lane & 15);
  int k0 = kt * 32 + (lane >> 4) * 8;
  union { uint4 q[2]; u16 h[16]; } pk;
#pragma unroll
  for (int i = 0; i < 16; ++i) pk.h[i] = f2bfu(w[(size_t)(k0 + (i < 8 ? i : 8 + i)) * N + n]);
  uint4* dst = (uint4*)(out + (size_t)tile * 512 + lane * 16);
  dst[0] = pk.q[0]; dst[1] = pk.q[1];
}

// ================= K0: row layernorms =================
// aln: row-major (read per-element later); sln/sib: A-swizzled (GEMM A operands)
__global__ void k_ln_rows(const float* __restrict__ a_i, const float* __restrict__ s_i,
                          const float* __restrict__ lnsw, const float* __restrict__ lnsb,
                          u16* __restrict__ aln, u16* __restrict__ sln, u16* __restrict__ si) {
  __shared__ float red[256];
  int row = blockIdx.x, t = threadIdx.x;
  {
    const float* ar = a_i + (size_t)row * CA;
    float s = 0.f, ss = 0.f;
    for (int c = t; c < CA; c += 256) { float x = ar[c]; s += x; ss += x * x; }
    red[t] = s; __syncthreads();
    for (int o = 128; o > 0; o >>= 1) { if (t < o) red[t] += red[t + o]; __syncthreads(); }
    float mean = red[0] / CA; __syncthreads();
    red[t] = ss; __syncthreads();
    for (int o = 128; o > 0; o >>= 1) { if (t < o) red[t] += red[t + o]; __syncthreads(); }
    float inv = rsqrtf(red[0] / CA - mean * mean + EPSf); __syncthreads();
    for (int c = t; c < CA; c += 256)
      aln[(size_t)row * CA + c] = f2bfu((ar[c] - mean) * inv);
  }
  {
    const float* sr = s_i + (size_t)row * CS;
    float s = 0.f, ss = 0.f;
    for (int c = t; c < CS; c += 256) { float x = sr[c]; s += x; ss += x * x; }
    red[t] = s; __syncthreads();
    for (int o = 128; o > 0; o >>= 1) { if (t < o) red[t] += red[t + o]; __syncthreads(); }
    float mean = red[0] / CS; __syncthreads();
    red[t] = ss; __syncthreads();
    for (int o = 128; o > 0; o >>= 1) { if (t < o) red[t] += red[t + o]; __syncthreads(); }
    float inv = rsqrtf(red[0] / CS - mean * mean + EPSf); __syncthreads();
    for (int c = t; c < CS; c += 256) {
      float y = (sr[c] - mean) * inv * lnsw[c] + lnsb[c];
      sln[aswz_idx(row, c, CS)] = f2bfu(y);
      si [aswz_idx(row, c, CS)] = f2bfu(sr[c]);
    }
  }
}

// ================= K1: AdaLN dual GEMM, LDS-free =================
// grid (M/16, N/128), block 256: wave w owns n-tile blockIdx.y*8+w
__global__ void k_adaln(const u16* __restrict__ sln, const u16* __restrict__ wWs,
                        const u16* __restrict__ wWnb, const float* __restrict__ bs,
                        const u16* __restrict__ aln, u16* __restrict__ abuf) {
  int t = threadIdx.x, wave = t >> 5, lane = t & 31;
  int mt = blockIdx.x, nt = blockIdx.y * 8 + wave;
  const u16* aT  = sln  + (size_t)mt * (CS >> 5) * 512;
  const u16* b0T = wWs  + (size_t)nt * (CS >> 5) * 512;
  const u16* b1T = wWnb + (size_t)nt * (CS >> 5) * 512;
  v8f acc0 = {}, acc1 = {};
#pragma unroll 4
  for (int kt = 0; kt < (CS >> 5); ++kt) {
    if (kt + 2 < (CS >> 5)) __builtin_prefetch(b0T + (size_t)(kt + 2) * 512, 0, 1);
    v16bf a = load_frag_sw(aT + (size_t)kt * 512);
    acc0 = wmma_bf(a, load_frag_sw(b0T + (size_t)kt * 512), acc0);
    acc1 = wmma_bf(a, load_frag_sw(b1T + (size_t)kt * 512), acc1);
  }
  int n = nt * 16 + (lane & 15);
  int mh = (lane >> 4) * 8;
#pragma unroll
  for (int r = 0; r < 8; ++r) {
    int m = mt * 16 + mh + r;
    float g = sigf(acc0[r] + bs[n]);
    float res = g * bfu2f(aln[(size_t)m * CA + n]) + acc1[r];
    abuf[aswz_idx(m, n, CA)] = f2bfu(res);
  }
}

// ================= K2: projection GEMM, LDS-free -> [B,H,I,D] =================
__global__ void k_proj(const u16* __restrict__ abuf, const u16* __restrict__ w,
                       const float* __restrict__ bias, int do_sig, u16* __restrict__ out) {
  int t = threadIdx.x, wave = t >> 5, lane = t & 31;
  int mt = blockIdx.x, nt = blockIdx.y * 8 + wave;
  const u16* aT = abuf + (size_t)mt * (CA >> 5) * 512;
  const u16* bT = w    + (size_t)nt * (CA >> 5) * 512;
  v8f acc = {};
#pragma unroll 4
  for (int kt = 0; kt < (CA >> 5); ++kt) {
    if (kt + 2 < (CA >> 5)) __builtin_prefetch(bT + (size_t)(kt + 2) * 512, 0, 1);
    acc = wmma_bf(load_frag_sw(aT + (size_t)kt * 512),
                  load_frag_sw(bT + (size_t)kt * 512), acc);
  }
  int n = nt * 16 + (lane & 15);
  int mh = (lane >> 4) * 8;
  int h = n / DD, d = n % DD;
#pragma unroll
  for (int r = 0; r < 8; ++r) {
    int m = mt * 16 + mh + r;
    int b = m / II, i = m % II;
    float v = acc[r] + (bias ? bias[n] : 0.0f);
    if (do_sig) v = sigf(v);
    out[(((size_t)b * HH + h) * II + i) * (size_t)DD + d] = f2bfu(v);
  }
}

// ================= K3: pair bias — LN(z) x Wb + beta -> [B,H,I,J] =================
__global__ void k_pairbias(const float* __restrict__ z, const float* __restrict__ beta,
                           const float* __restrict__ lw, const float* __restrict__ lb,
                           const u16* __restrict__ wWb, u16* __restrict__ bias) {
  __shared__ __align__(16) u16 Zs[8][16 * 128];
  int t = threadIdx.x, wave = t >> 5, lane = t & 31;
  size_t row0 = (size_t)blockIdx.x * 128 + (size_t)wave * 16;
  int lr = lane >> 1, lh = lane & 1;          // 2 lanes per row, 64 ch each
  const float4* zv = (const float4*)(z + (row0 + lr) * (size_t)CZ + lh * 64);
  float s = 0.f, ss = 0.f;
  float vals[64];
#pragma unroll
  for (int c4 = 0; c4 < 16; ++c4) {
    float4 x = zv[c4];
    vals[c4 * 4 + 0] = x.x; vals[c4 * 4 + 1] = x.y;
    vals[c4 * 4 + 2] = x.z; vals[c4 * 4 + 3] = x.w;
    s += x.x + x.y + x.z + x.w;
    ss += x.x * x.x + x.y * x.y + x.z * x.z + x.w * x.w;
  }
  s  += __shfl_xor(s, 1);
  ss += __shfl_xor(ss, 1);
  float mean = s / CZ;
  float inv = rsqrtf(ss / CZ - mean * mean + EPSf);
#pragma unroll
  for (int cb = 0; cb < 64; cb += 8) {
    union { uint4 q; u16 h[8]; } pk;
#pragma unroll
    for (int c = 0; c < 8; ++c) {
      int k = lh * 64 + cb + c;
      pk.h[c] = f2bfu((vals[cb + c] - mean) * inv * lw[k] + lb[k]);
    }
    *(uint4*)&Zs[wave][lr * 128 + lh * 64 + cb] = pk.q;
  }
  __syncthreads();
  v8f acc = {};
#pragma unroll
  for (int kt = 0; kt < (CZ >> 5); ++kt)
    acc = wmma_bf(load_frag_a_lds(&Zs[wave][kt * 32], 128),
                  load_frag_sw(wWb + (size_t)kt * 512), acc);
  int h = lane & 15, mh = (lane >> 4) * 8;
  size_t rr0 = row0 + mh;                     // 8 consecutive flat (b,i,j) rows
  union { uint4 q; u16 hh[8]; } pk;
#pragma unroll
  for (int r = 0; r < 8; ++r) pk.hh[r] = f2bfu(acc[r] + beta[(rr0 + r) * HH + h]);
  size_t j0 = rr0 % II, bi = rr0 / II, i = bi % II, b = bi / II;
  *(uint4*)&bias[(((b * HH + h) * II + i) * (size_t)II) + j0] = pk.q;
}

// ================= K4: flash attention per (b,h) =================
__global__ void k_attn(const u16* __restrict__ qb, const u16* __restrict__ kb_,
                       const u16* __restrict__ vb_, const u16* __restrict__ gb,
                       const u16* __restrict__ bias, u16* __restrict__ obuf) {
  __shared__ __align__(16) u16 Ks[16 * 48];
  __shared__ __align__(16) u16 Vs[16 * 48];
  __shared__ __align__(16) u16 Ps[8][16 * 16];
  int t = threadIdx.x, wave = t >> 5, lane = t & 31;
  int bh = blockIdx.y, b = bh >> 4, h = bh & 15;
  int i0 = blockIdx.x * 128 + wave * 16;
  size_t base = (size_t)bh * II * DD;
  int mq = lane & 15, half = lane >> 4;
  v16bf qa0, qa1;
#pragma unroll
  for (int v = 0; v < 8; ++v) {
#pragma unroll
    for (int e = 0; e < 2; ++e) {
      int k = (v < 4 ? 0 : 16) + half * 8 + (v & 3) * 2 + e;
      qa0[2 * v + e] = u2b(qb[base + (size_t)(i0 + mq) * DD + k]);
      qa1[2 * v + e] = (k < 16) ? u2b(qb[base + (size_t)(i0 + mq) * DD + 32 + k]) : u2b(0);
    }
  }
  float rowm[8], rowl[8];
#pragma unroll
  for (int r = 0; r < 8; ++r) { rowm[r] = -3.0e38f; rowl[r] = 0.0f; }
  v8f o0 = {}, o1 = {}, o2 = {};
  const float scale = 0.14433756729740643f;   // 1/sqrt(48)
  int mh = half * 8;
  for (int j0 = 0; j0 < II; j0 += 16) {
    __syncthreads();
    // vectorized K/V staging: 96 uint4 each
    if (t < 96)
      ((uint4*)Ks)[t] = ((const uint4*)(kb_ + base + (size_t)j0 * DD))[t];
    else if (t < 192)
      ((uint4*)Vs)[t - 96] = ((const uint4*)(vb_ + base + (size_t)j0 * DD))[t - 96];
    __syncthreads();
    v16bf kf0, kf1;
#pragma unroll
    for (int v = 0; v < 8; ++v) {
#pragma unroll
      for (int e = 0; e < 2; ++e) {
        int d0 = (v < 4 ? 0 : 16) + half * 8 + (v & 3) * 2 + e;
        kf0[2 * v + e] = u2b(Ks[mq * 48 + d0]);
        int d1 = 32 + d0;
        kf1[2 * v + e] = (d0 < 16) ? u2b(Ks[mq * 48 + d1]) : u2b(0);
      }
    }
    v8f sc = {};
    sc = wmma_bf(qa0, kf0, sc);
    sc = wmma_bf(qa1, kf1, sc);
    int jcol = j0 + (lane & 15);
    float pv[8];
#pragma unroll
    for (int r = 0; r < 8; ++r)
      pv[r] = sc[r] * scale +
              bfu2f(bias[((size_t)bh * II + (i0 + mh + r)) * (size_t)II + jcol]);
#pragma unroll
    for (int r = 0; r < 8; ++r) {
      float cm = pv[r];
      for (int o = 1; o < 16; o <<= 1) cm = fmaxf(cm, __shfl_xor(cm, o));
      float nm = fmaxf(rowm[r], cm);
      float so = __expf(rowm[r] - nm);
      float p  = __expf(pv[r] - nm);
      float ps = p;
      for (int o = 1; o < 16; o <<= 1) ps += __shfl_xor(ps, o);
      rowl[r] = rowl[r] * so + ps;
      rowm[r] = nm;
      o0[r] *= so; o1[r] *= so; o2[r] *= so;
      Ps[wave][(mh + r) * 16 + (lane & 15)] = f2bfu(p);
    }
    v16bf pa;
#pragma unroll
    for (int v = 0; v < 8; ++v) {
#pragma unroll
      for (int e = 0; e < 2; ++e) {
        int k = (v < 4 ? 0 : 16) + half * 8 + (v & 3) * 2 + e;
        pa[2 * v + e] = (k < 16) ? u2b(Ps[wave][mq * 16 + k]) : u2b(0);
      }
    }
#pragma unroll
    for (int dt = 0; dt < 3; ++dt) {
      v16bf vf;
#pragma unroll
      for (int v = 0; v < 8; ++v) {
#pragma unroll
        for (int e = 0; e < 2; ++e) {
          int jl = (v < 4 ? 0 : 16) + half * 8 + (v & 3) * 2 + e;
          vf[2 * v + e] = (jl < 16) ? u2b(Vs[jl * 48 + dt * 16 + mq]) : u2b(0);
        }
      }
      if (dt == 0) o0 = wmma_bf(pa, vf, o0);
      else if (dt == 1) o1 = wmma_bf(pa, vf, o1);
      else o2 = wmma_bf(pa, vf, o2);
    }
  }
  // epilogue: normalize, gate, store A-swizzled for k_out
#pragma unroll
  for (int dt = 0; dt < 3; ++dt) {
#pragma unroll
    for (int r = 0; r < 8; ++r) {
      int i = i0 + mh + r;
      int d = dt * 16 + (lane & 15);
      float v = (dt == 0 ? o0[r] : dt == 1 ? o1[r] : o2[r]) / rowl[r];
      v *= bfu2f(gb[base + (size_t)i * DD + d]);
      obuf[aswz_idx(b * II + i, h * DD + d, HCd)] = f2bfu(v);
    }
  }
}

// ================= K5: out = sigmoid(si x Ws_out + bs_out) * (O x Wo), LDS-free =================
__global__ void k_out(const u16* __restrict__ obuf, const u16* __restrict__ wWo,
                      const u16* __restrict__ si, const u16* __restrict__ wWso,
                      const float* __restrict__ bs_out, float* __restrict__ out) {
  int t = threadIdx.x, wave = t >> 5, lane = t & 31;
  int mt = blockIdx.x, nt = blockIdx.y * 8 + wave;
  v8f acc1 = {};
  {
    const u16* aT = obuf + (size_t)mt * (HCd >> 5) * 512;
    const u16* bT = wWo  + (size_t)nt * (HCd >> 5) * 512;
#pragma unroll 4
    for (int kt = 0; kt < (HCd >> 5); ++kt) {
      if (kt + 2 < (HCd >> 5)) __builtin_prefetch(bT + (size_t)(kt + 2) * 512, 0, 1);
      acc1 = wmma_bf(load_frag_sw(aT + (size_t)kt * 512),
                     load_frag_sw(bT + (size_t)kt * 512), acc1);
    }
  }
  v8f acc2 = {};
  {
    const u16* aT = si   + (size_t)mt * (CS >> 5) * 512;
    const u16* bT = wWso + (size_t)nt * (CS >> 5) * 512;
#pragma unroll 4
    for (int kt = 0; kt < (CS >> 5); ++kt) {
      acc2 = wmma_bf(load_frag_sw(aT + (size_t)kt * 512),
                     load_frag_sw(bT + (size_t)kt * 512), acc2);
    }
  }
  int n = nt * 16 + (lane & 15);
  int mh = (lane >> 4) * 8;
#pragma unroll
  for (int r = 0; r < 8; ++r) {
    int m = mt * 16 + mh + r;
    out[(size_t)m * CA + n] = sigf(acc2[r] + bs_out[n]) * acc1[r];
  }
}

// ================= host launcher =================
extern "C" void kernel_launch(void* const* d_in, const int* in_sizes, int n_in,
                              void* d_out, int out_size, void* d_ws, size_t ws_size,
                              hipStream_t stream) {
  const float* a_i      = (const float*)d_in[0];
  const float* s_i      = (const float*)d_in[1];
  const float* z_ij     = (const float*)d_in[2];
  const float* beta_ij  = (const float*)d_in[3];
  const float* lns_w    = (const float*)d_in[4];
  const float* lns_b    = (const float*)d_in[5];
  const float* adaln_Ws = (const float*)d_in[6];
  const float* adaln_bs = (const float*)d_in[7];
  const float* adaln_Wnb= (const float*)d_in[8];
  const float* Wq       = (const float*)d_in[9];
  const float* bq       = (const float*)d_in[10];
  const float* Wk       = (const float*)d_in[11];
  const float* Wv       = (const float*)d_in[12];
  const float* lnb_w    = (const float*)d_in[13];
  const float* lnb_b    = (const float*)d_in[14];
  const float* Wb       = (const float*)d_in[15];
  const float* Wg       = (const float*)d_in[16];
  const float* Wo       = (const float*)d_in[17];
  const float* Ws_out   = (const float*)d_in[18];
  const float* bs_out   = (const float*)d_in[19];
  float* out = (float*)d_out;

  char* ws = (char*)d_ws;
  size_t off = 0;
  auto alloc = [&](size_t bytes) -> char* {
    char* p = ws + off;
    off += (bytes + 255) & ~(size_t)255;
    return p;
  };
  u16* aln  = (u16*)alloc((size_t)MROWS * CA * 2);
  u16* sln  = (u16*)alloc((size_t)MROWS * CS * 2);
  u16* sib  = (u16*)alloc((size_t)MROWS * CS * 2);
  u16* abuf = (u16*)alloc((size_t)MROWS * CA * 2);
  u16* qb   = (u16*)alloc((size_t)MROWS * HCd * 2);
  u16* kb   = (u16*)alloc((size_t)MROWS * HCd * 2);
  u16* vb   = (u16*)alloc((size_t)MROWS * HCd * 2);
  u16* gb   = (u16*)alloc((size_t)MROWS * HCd * 2);
  u16* bias = (u16*)alloc((size_t)Bb * HH * II * II * 2);
  u16* obuf = (u16*)alloc((size_t)MROWS * HCd * 2);
  u16* wWs  = (u16*)alloc((size_t)CS * CA * 2);
  u16* wWnb = (u16*)alloc((size_t)CS * CA * 2);
  u16* wWq  = (u16*)alloc((size_t)CA * HCd * 2);
  u16* wWk  = (u16*)alloc((size_t)CA * HCd * 2);
  u16* wWv  = (u16*)alloc((size_t)CA * HCd * 2);
  u16* wWg  = (u16*)alloc((size_t)CA * HCd * 2);
  u16* wWo  = (u16*)alloc((size_t)HCd * CA * 2);
  u16* wWso = (u16*)alloc((size_t)CS * CA * 2);
  u16* wWb  = (u16*)alloc((size_t)CZ * HH * 2);

  auto cvt = [&](const float* src, u16* dst, int K, int N) {
    int n_units = (K / 32) * (N / 16) * 32;
    k_cvt_bswz<<<(n_units + 255) / 256, 256, 0, stream>>>(src, dst, K, N, n_units);
  };
  cvt(adaln_Ws,  wWs,  CS, CA);
  cvt(adaln_Wnb, wWnb, CS, CA);
  cvt(Wq, wWq, CA, HCd);
  cvt(Wk, wWk, CA, HCd);
  cvt(Wv, wWv, CA, HCd);
  cvt(Wg, wWg, CA, HCd);
  cvt(Wo, wWo, HCd, CA);
  cvt(Ws_out, wWso, CS, CA);
  cvt(Wb, wWb, CZ, HH);

  k_ln_rows<<<MROWS, 256, 0, stream>>>(a_i, s_i, lns_w, lns_b, aln, sln, sib);

  k_adaln<<<dim3(MROWS / 16, CA / 128), 256, 0, stream>>>(sln, wWs, wWnb, adaln_bs, aln, abuf);

  k_proj<<<dim3(MROWS / 16, HCd / 128), 256, 0, stream>>>(abuf, wWq, bq,      0, qb);
  k_proj<<<dim3(MROWS / 16, HCd / 128), 256, 0, stream>>>(abuf, wWk, nullptr, 0, kb);
  k_proj<<<dim3(MROWS / 16, HCd / 128), 256, 0, stream>>>(abuf, wWv, nullptr, 0, vb);
  k_proj<<<dim3(MROWS / 16, HCd / 128), 256, 0, stream>>>(abuf, wWg, nullptr, 1, gb);

  k_pairbias<<<(Bb * II * II) / 128, 256, 0, stream>>>(z_ij, beta_ij, lnb_w, lnb_b, wWb, bias);

  k_attn<<<dim3(II / 128, Bb * HH), 256, 0, stream>>>(qb, kb, vb, gb, bias, obuf);

  k_out<<<dim3(MROWS / 16, CA / 128), 256, 0, stream>>>(obuf, wWo, sib, wWso, bs_out, out);

  (void)in_sizes; (void)n_in; (void)out_size; (void)ws_size;
}
